// DistributedAFNO2Dv2_29592324670088
// MI455X (gfx1250) — compile-verified, matrix operand order checked by reading, not run on hardware
//
#include <hip/hip_runtime.h>
#include <hip/hip_bf16.h>

// ---------------------------------------------------------------------------
// AFNO2D forward for MI455X (gfx1250, wave32)
//   x:(2,768,256,256) f32 -> rfft2(ortho) -> per-block complex MLP (WMMA f16)
//   -> softshrink -> irfft2(ortho) -> + x
// Column FFT passes use the Tensor Data Mover (TDM) for the strided
// column gather/scatter (stride 129*8B), tracked with TENSORcnt.
// ---------------------------------------------------------------------------

typedef __attribute__((ext_vector_type(16))) _Float16 v16h;
typedef __attribute__((ext_vector_type(8)))  float    v8f;
typedef __attribute__((ext_vector_type(4)))  unsigned int u32x4;
typedef __attribute__((ext_vector_type(4)))  int          i32x4;
typedef __attribute__((ext_vector_type(8)))  int          i32x8;

namespace afno {
constexpr int Bn  = 2;
constexpr int Cn  = 768;
constexpr int Hn  = 256;
constexpr int Wn  = 256;
constexpr int Wf  = 129;            // W/2 + 1
constexpr int NB  = 8;
constexpr int BS  = 96;             // channels per block
constexpr int M   = Hn * Wf;        // 33024 frequency points per channel
constexpr int TM  = 64;             // M-tile per workgroup (4 waves x 16)
constexpr int NT  = M / TM;         // 516 tiles per (b,k)
constexpr float LAM   = 0.01f;
constexpr float SC16  = 0.0625f;    // 1/sqrt(256) per 1-D ortho pass
}

#if defined(__gfx1250__) && __has_builtin(__builtin_amdgcn_tensor_load_to_lds) && \
    __has_builtin(__builtin_amdgcn_s_wait_tensorcnt)
#define AFNO_TDM_LOAD 1
#else
#define AFNO_TDM_LOAD 0
#endif
#if AFNO_TDM_LOAD && __has_builtin(__builtin_amdgcn_tensor_store_from_lds)
#define AFNO_TDM_STORE 1
#else
#define AFNO_TDM_STORE 0
#endif

// ---------------------------------------------------------------------------
// Build a 2D TDM descriptor: one column of Hn float2 elements (8B each),
// element stride Wf, into a contiguous 2KB LDS region.
//   g0: count=1 | lds_addr | global_addr[56:0] | type=2
//   g1: data_size=8B, tensor_dim0=Wf, tensor_dim1=Hn, tile_dim0=1,
//       tile_dim1=Hn, tensor_dim0_stride=Wf
// ---------------------------------------------------------------------------
__device__ __forceinline__ void tdm_col_desc(unsigned long long gaddr,
                                             unsigned lds_byte,
                                             u32x4& g0, i32x8& g1) {
  g0[0] = 1u;                                       // count = 1 (valid D#)
  g0[1] = lds_byte;                                 // LDS byte address
  g0[2] = (unsigned)gaddr;                          // global addr [31:0]
  g0[3] = (unsigned)((gaddr >> 32) & 0x01FFFFFFull) // global addr [56:32]
        | (2u << 30);                               // type = 2 (image)
  g1[0] = (int)(3u << 16);                          // data_size = 3 (8 bytes)
  g1[1] = (int)((unsigned)afno::Wf << 16);          // tensor_dim0[15:0]
  g1[2] = (int)((unsigned)afno::Hn << 16);          // tensor_dim1[15:0]
  g1[3] = (int)(1u << 16);                          // tile_dim0 = 1
  g1[4] = (int)afno::Hn;                            // tile_dim1 = 256
  g1[5] = (int)afno::Wf;                            // tensor_dim0_stride[31:0]
  g1[6] = 0;
  g1[7] = 0;
}

// ---------------------------------------------------------------------------
// 256-point in-LDS radix-2 FFT (DIT, input already bit-reversed).
// One wave owns one 256-entry float2 region; __syncthreads() per stage.
// sign = -1 forward, +1 inverse.
// ---------------------------------------------------------------------------
__device__ __forceinline__ void fft256_stages(float2* s, int lane, float sign) {
  #pragma unroll
  for (int st = 1; st <= 8; ++st) {
    const int mlen = 1 << st;
    const int hf   = mlen >> 1;
    for (int bfy = lane; bfy < 128; bfy += 32) {
      const int grp = bfy >> (st - 1);
      const int pos = bfy & (hf - 1);
      const int i0  = (grp << st) + pos;
      const int i1  = i0 + hf;
      const float ang = sign * 6.28318530717958647f * (float)pos / (float)mlen;
      const float c = __cosf(ang);
      const float sn = __sinf(ang);
      const float2 u = s[i0];
      const float2 v = s[i1];
      float2 t; t.x = v.x * c - v.y * sn; t.y = v.x * sn + v.y * c;
      float2 a; a.x = u.x + t.x; a.y = u.y + t.y;
      float2 b; b.x = u.x - t.x; b.y = u.y - t.y;
      s[i0] = a;
      s[i1] = b;
    }
    __syncthreads();
  }
}

__device__ __forceinline__ int bitrev8(int j) { return (int)(__brev((unsigned)j) >> 24); }

// ---------------------------------------------------------------------------
// Kernel 1: per-row real FFT along W, keep 129 bins, scale 1/16.
// One wave per row; 8 rows per 256-thread block. rows = B*C*H = 393216.
// ---------------------------------------------------------------------------
__global__ __launch_bounds__(256) void afno_row_rfft(const float* __restrict__ x,
                                                     float2* __restrict__ spec) {
  __shared__ float2 buf[8 * 256];
  const int wave = threadIdx.x >> 5;
  const int lane = threadIdx.x & 31;
  const int row  = blockIdx.x * 8 + wave;             // (b*C + c)*H + h
  float2* s = &buf[wave * 256];
  const float* xr = x + (size_t)row * afno::Wn;
  for (int j = lane; j < 256; j += 32) {
    float2 v; v.x = xr[j]; v.y = 0.0f;
    s[bitrev8(j)] = v;
  }
  __syncthreads();
  fft256_stages(s, lane, -1.0f);
  float2* so = spec + (size_t)row * afno::Wf;
  for (int j = lane; j < afno::Wf; j += 32) {
    float2 v = s[j];
    v.x *= afno::SC16; v.y *= afno::SC16;
    so[j] = v;
  }
}

// ---------------------------------------------------------------------------
// Kernel 2/4: complex FFT along H for each (b,c,wf) column, in place,
// scale 1/16. cols = B*C*Wf = 198144. sign = -1 fwd, +1 inv.
// Column gather/scatter via TDM (stride Wf*8 bytes), one descriptor per wave.
// ---------------------------------------------------------------------------
__global__ __launch_bounds__(256) void afno_col_fft(float2* __restrict__ spec, float sign) {
  __shared__ float2 buf[8 * 256];
  const int wave = threadIdx.x >> 5;
  const int lane = threadIdx.x & 31;
  const int col  = blockIdx.x * 8 + wave;
  const int wf   = col % afno::Wf;
  const int bc   = col / afno::Wf;
  float2* s = &buf[wave * 256];
  const size_t base = (size_t)bc * afno::M + wf;      // stride Wf between h's

#if AFNO_TDM_LOAD
  const unsigned long long ga = (unsigned long long)(uintptr_t)(spec + base);
  const unsigned ldsb = (unsigned)(uintptr_t)(void*)s; // LDS aperture: addr[31:0]
  u32x4 g0; i32x8 g1;
  tdm_col_desc(ga, ldsb, g0, g1);
  const i32x4 z4 = {0, 0, 0, 0};
#if __clang_major__ >= 23
  const i32x8 z8 = {0, 0, 0, 0, 0, 0, 0, 0};
  __builtin_amdgcn_tensor_load_to_lds(g0, g1, z4, z4, z8, 0);
#else
  __builtin_amdgcn_tensor_load_to_lds(g0, g1, z4, z4, 0);
#endif
  __builtin_amdgcn_s_wait_tensorcnt(0);               // own tile landed
#else
  for (int j = lane; j < 256; j += 32)
    s[j] = spec[base + (size_t)j * afno::Wf];
#endif

  // In-place bit-reversal permutation (disjoint pairs, each handled once).
  for (int j = lane; j < 256; j += 32) {
    const int r = bitrev8(j);
    if (j < r) { const float2 t = s[j]; s[j] = s[r]; s[r] = t; }
  }
  __syncthreads();
  fft256_stages(s, lane, sign);

  // Scale in LDS (own region only), then scatter back.
  for (int j = lane; j < 256; j += 32) {
    float2 v = s[j];
    v.x *= afno::SC16; v.y *= afno::SC16;
    s[j] = v;
  }
#if AFNO_TDM_STORE
  asm volatile("s_wait_dscnt 0x0" ::: "memory");      // LDS writes visible to TDM
#if __clang_major__ >= 23
  {
    const i32x8 z8s = {0, 0, 0, 0, 0, 0, 0, 0};
    __builtin_amdgcn_tensor_store_from_lds(g0, g1, z4, z4, z8s, 0);
  }
#else
  __builtin_amdgcn_tensor_store_from_lds(g0, g1, z4, z4, 0);
#endif
  __builtin_amdgcn_s_wait_tensorcnt(0);
#else
  for (int j = lane; j < 256; j += 32)
    spec[base + (size_t)j * afno::Wf] = s[j];
#endif
}

// ---------------------------------------------------------------------------
// WMMA helpers (confirmed builtin: v_wmma_f32_16x16x32_f16, f32 accumulate).
// ---------------------------------------------------------------------------
__device__ __forceinline__ v8f wmma16(v16h a, v16h b, v8f c) {
  return __builtin_amdgcn_wmma_f32_16x16x32_f16(false, a, false, b, (short)0, c,
                                                false, false);
}

__device__ __forceinline__ int wmma_k_of(int r, int h, int laneHi) {
  const int base = (r < 4) ? (2 * r) : (16 + 2 * (r - 4));
  return base + h + (laneHi ? 8 : 0);
}

__device__ __forceinline__ v16h load_A(const _Float16* s, int mbase, int kbase,
                                       int lane) {
  const int m = mbase + (lane & 15);
  const int laneHi = (lane >> 4) & 1;
  v16h a;
  #pragma unroll
  for (int r = 0; r < 8; ++r) {
    const int K = kbase + wmma_k_of(r, 0, laneHi);
    a[2 * r]     = s[m * afno::BS + K];
    a[2 * r + 1] = s[m * afno::BS + K + 1];
  }
  return a;
}

__device__ __forceinline__ void load_B(const float* __restrict__ w, int k,
                                       int kbase, int otile, int lane,
                                       v16h& bre, v16h& bim) {
  const int n = otile * 16 + (lane & 15);
  const int laneHi = (lane >> 4) & 1;
  #pragma unroll
  for (int r = 0; r < 8; ++r) {
    const int K = kbase + wmma_k_of(r, 0, laneHi);
    const float* p0 = &w[(((size_t)k * afno::BS + K) * afno::BS + n) * 2];
    const float* p1 = &w[(((size_t)k * afno::BS + K + 1) * afno::BS + n) * 2];
    bre[2 * r]     = (_Float16)p0[0];
    bre[2 * r + 1] = (_Float16)p1[0];
    bim[2 * r]     = (_Float16)p0[1];
    bim[2 * r + 1] = (_Float16)p1[1];
  }
}

__device__ __forceinline__ float softshrink(float v) {
  float a = fabsf(v) - afno::LAM;
  a = a > 0.0f ? a : 0.0f;
  return v > 0.0f ? a : (v < 0.0f ? -a : 0.0f);
}

// ---------------------------------------------------------------------------
// Kernel 3: frequency-domain block MLP, in place on spec.
// Block = 128 threads (4 waves), each wave owns a 16-row M-tile of the 64-row
// block tile. Complex GEMM: re += A_re*W_re + A_im*(-W_im);
//                            im += A_re*W_im + A_im* W_re.
// grid = B * NB * NT = 8256.
// ---------------------------------------------------------------------------
__global__ __launch_bounds__(128) void afno_freq_mlp(float2* __restrict__ spec,
                                                     const float* __restrict__ w1,
                                                     const float* __restrict__ b1,
                                                     const float* __restrict__ w2,
                                                     const float* __restrict__ b2) {
  using namespace afno;
  __shared__ _Float16 sRe[TM * BS];   // [m][i] activation / o1 plane (real)
  __shared__ _Float16 sIm[TM * BS];   // [m][i] activation / o1 plane (imag)

  const int tid  = threadIdx.x;
  const int lane = tid & 31;
  const int wave = tid >> 5;
  const int laneHi = (lane >> 4) & 1;
  const int n0     = lane & 15;
  const int m_off  = wave * 16;

  int t = blockIdx.x;
  const int tile = t % NT; t /= NT;
  const int k    = t % NB; t /= NB;
  const int b    = t;
  const int xy0  = tile * TM;

  // ---- stage X tile (f32 global -> f16 LDS planes), coalesced along xy ----
  for (int e = tid; e < BS * TM; e += 128) {
    const int i = e >> 6;          // channel within block
    const int m = e & (TM - 1);    // point within tile
    const size_t gi = ((size_t)(b * Cn + k * BS + i)) * M + xy0 + m;
    const float2 v = spec[gi];
    sRe[m * BS + i] = (_Float16)v.x;
    sIm[m * BS + i] = (_Float16)v.y;
    __builtin_prefetch(&spec[gi + TM], 0, 1);
  }
  __syncthreads();

  v8f accRe[6], accIm[6];

  // ============================ layer 1 ====================================
  #pragma unroll
  for (int nt = 0; nt < 6; ++nt) {
    const float br = b1[((size_t)k * BS + nt * 16 + n0) * 2 + 0];
    const float bi = b1[((size_t)k * BS + nt * 16 + n0) * 2 + 1];
    #pragma unroll
    for (int e = 0; e < 8; ++e) { accRe[nt][e] = br; accIm[nt][e] = bi; }
  }
  #pragma unroll
  for (int kc = 0; kc < 3; ++kc) {
    const v16h aRe = load_A(sRe, m_off, kc * 32, lane);
    const v16h aIm = load_A(sIm, m_off, kc * 32, lane);
    #pragma unroll
    for (int nt = 0; nt < 6; ++nt) {
      v16h bre, bim;
      load_B(w1, k, kc * 32, nt, lane, bre, bim);
      const v16h bimNeg = -bim;
      accRe[nt] = wmma16(aRe, bre,    accRe[nt]);
      accRe[nt] = wmma16(aIm, bimNeg, accRe[nt]);
      accIm[nt] = wmma16(aRe, bim,    accIm[nt]);
      accIm[nt] = wmma16(aIm, bre,    accIm[nt]);
    }
  }
  __syncthreads();
  #pragma unroll
  for (int nt = 0; nt < 6; ++nt) {
    const int o = nt * 16 + n0;
    #pragma unroll
    for (int r = 0; r < 8; ++r) {
      const int m = m_off + r + 8 * laneHi;
      sRe[m * BS + o] = (_Float16)fmaxf(accRe[nt][r], 0.0f);
      sIm[m * BS + o] = (_Float16)fmaxf(accIm[nt][r], 0.0f);
    }
  }
  __syncthreads();

  // ============================ layer 2 ====================================
  #pragma unroll
  for (int nt = 0; nt < 6; ++nt) {
    const float br = b2[((size_t)k * BS + nt * 16 + n0) * 2 + 0];
    const float bi = b2[((size_t)k * BS + nt * 16 + n0) * 2 + 1];
    #pragma unroll
    for (int e = 0; e < 8; ++e) { accRe[nt][e] = br; accIm[nt][e] = bi; }
  }
  #pragma unroll
  for (int kc = 0; kc < 3; ++kc) {
    const v16h aRe = load_A(sRe, m_off, kc * 32, lane);
    const v16h aIm = load_A(sIm, m_off, kc * 32, lane);
    #pragma unroll
    for (int nt = 0; nt < 6; ++nt) {
      v16h bre, bim;
      load_B(w2, k, kc * 32, nt, lane, bre, bim);
      const v16h bimNeg = -bim;
      accRe[nt] = wmma16(aRe, bre,    accRe[nt]);
      accRe[nt] = wmma16(aIm, bimNeg, accRe[nt]);
      accIm[nt] = wmma16(aRe, bim,    accIm[nt]);
      accIm[nt] = wmma16(aIm, bre,    accIm[nt]);
    }
  }

  // ---- softshrink + in-place writeback ------------------------------------
  #pragma unroll
  for (int nt = 0; nt < 6; ++nt) {
    const int o = nt * 16 + n0;
    #pragma unroll
    for (int r = 0; r < 8; ++r) {
      const int m = m_off + r + 8 * laneHi;
      float2 v;
      v.x = softshrink(accRe[nt][r]);
      v.y = softshrink(accIm[nt][r]);
      spec[((size_t)(b * Cn + k * BS + o)) * M + xy0 + m] = v;
    }
  }
}

// ---------------------------------------------------------------------------
// Kernel 5: per-row inverse real FFT (conjugate-symmetric extend), + residual x.
// ---------------------------------------------------------------------------
__global__ __launch_bounds__(256) void afno_row_irfft(const float2* __restrict__ spec,
                                                      const float* __restrict__ xin,
                                                      float* __restrict__ out) {
  __shared__ float2 buf[8 * 256];
  const int wave = threadIdx.x >> 5;
  const int lane = threadIdx.x & 31;
  const int row  = blockIdx.x * 8 + wave;
  float2* s = &buf[wave * 256];
  const float2* sp = spec + (size_t)row * afno::Wf;
  for (int j = lane; j < 256; j += 32) {
    float2 v;
    if (j < afno::Wf) {
      v = sp[j];
    } else {
      const float2 c = sp[256 - j];
      v.x = c.x; v.y = -c.y;
    }
    s[bitrev8(j)] = v;
  }
  __syncthreads();
  fft256_stages(s, lane, +1.0f);
  const size_t rg = (size_t)row * afno::Wn;
  for (int j = lane; j < 256; j += 32)
    out[rg + j] = s[j].x * afno::SC16 + xin[rg + j];
}

// ---------------------------------------------------------------------------
extern "C" void kernel_launch(void* const* d_in, const int* in_sizes, int n_in,
                              void* d_out, int out_size, void* d_ws, size_t ws_size,
                              hipStream_t stream) {
  (void)in_sizes; (void)n_in; (void)out_size; (void)ws_size;
  const float* x  = (const float*)d_in[0];
  const float* w1 = (const float*)d_in[1];
  const float* b1 = (const float*)d_in[2];
  const float* w2 = (const float*)d_in[3];
  const float* b2 = (const float*)d_in[4];
  float*  out  = (float*)d_out;
  float2* spec = (float2*)d_ws;   // (B,C,H,Wf) complex: 406 MB

  const int rowBlocks = (afno::Bn * afno::Cn * afno::Hn) / 8;   // 49152
  const int colBlocks = (afno::Bn * afno::Cn * afno::Wf) / 8;   // 24768
  const int mlpBlocks = afno::Bn * afno::NB * afno::NT;         // 8256

  afno_row_rfft <<<rowBlocks, 256, 0, stream>>>(x, spec);
  afno_col_fft  <<<colBlocks, 256, 0, stream>>>(spec, -1.0f);
  afno_freq_mlp <<<mlpBlocks, 128, 0, stream>>>(spec, w1, b1, w2, b2);
  afno_col_fft  <<<colBlocks, 256, 0, stream>>>(spec, +1.0f);
  afno_row_irfft<<<rowBlocks, 256, 0, stream>>>(spec, x, out);
}
